// MultiHeadAttention_29248727285944
// MI455X (gfx1250) — matrix operand order repaired and checked by reading, not compiled, not run on hardware
//
#include <hip/hip_runtime.h>
#include <hip/hip_bf16.h>

typedef __attribute__((ext_vector_type(16))) __bf16       bf16x16;
typedef __attribute__((ext_vector_type(8)))  float        f32x8;
typedef __attribute__((ext_vector_type(8)))  unsigned int u32x8;

#define B_SZ    16
#define L_SEQ   1025
#define L_PAD   1056              /* 33 * 32 : every 32-key tile is full   */
#define DMODEL  768
#define N_HEAD  12
#define D_HEAD  64
#define M_ROWS  (B_SZ * L_SEQ)    /* 16400 */
#define N_QKV   (3 * DMODEL)      /* 2304  */

__device__ __forceinline__ unsigned short f2bf(float f) {
  unsigned int u = __builtin_bit_cast(unsigned int, f);
  u += 0x7FFFu + ((u >> 16) & 1u);          // round-to-nearest-even
  return (unsigned short)(u >> 16);
}

__device__ __forceinline__ f32x8 wmma_bf16(const u32x8& a, const u32x8& b, f32x8 c) {
  return __builtin_amdgcn_wmma_f32_16x16x32_bf16(
      false, __builtin_bit_cast(bf16x16, a),
      false, __builtin_bit_cast(bf16x16, b),
      (short)0, c, false, false);
}

// K-pair index (even K) of element-pair v for the bf16 16x32 A-fragment.
__device__ __forceinline__ int a_kpair(int v, int hf) {
  return (v < 4) ? (2 * v + 8 * hf) : (16 + 2 * (v - 4) + 8 * hf);
}

// xor-lane shuffle via ds_swizzle (group-of-32, and_mask=0x1F, xor in [14:10])
#define SWIZ_XOR(m) (0x1F | ((m) << 10))
__device__ __forceinline__ float swz_max(float t, int imm_pattern_unused) { return t; }
template <int MASK>
__device__ __forceinline__ float swz_xor(float t) {
  return __builtin_bit_cast(float,
      __builtin_amdgcn_ds_swizzle(__builtin_bit_cast(int, t), SWIZ_XOR(MASK)));
}

// CDNA5 async global->LDS copy (ASYNCcnt path), 16B per lane.
__device__ __forceinline__ void async_load_b128(unsigned lds_byte_off,
                                                unsigned long long gaddr) {
  asm volatile("global_load_async_to_lds_b128 %0, %1, off"
               :: "v"(lds_byte_off), "v"(gaddr) : "memory");
}
__device__ __forceinline__ void wait_asynccnt0() {
  asm volatile("s_wait_asynccnt 0x0" ::: "memory");
}

// ---------------------------------------------------------------------------
// Tiled GEMM + bias: C[M,N] = A[M,K] @ B[K,N] + bias,  bf16 WMMA, f32 accum.
// 128x64 block tile, 8 waves x (16 rows x 64 cols), 4 WMMA per stage/wave.
// QKV_EPI: scatter into head-split Q/K/V bf16 buffers (interleaved head map).
// ---------------------------------------------------------------------------
template <bool A_BF16, bool QKV_EPI>
__global__ __launch_bounds__(256) void gemm_bias_wmma(
    const void* __restrict__ Aptr, const float* __restrict__ Bmat,
    const float* __restrict__ bias, float* __restrict__ Cout,
    unsigned short* __restrict__ Qb, unsigned short* __restrict__ Kb,
    unsigned short* __restrict__ Vb, int N, int K)
{
  __shared__ unsigned int As32[128 * 16];  // 128 rows x 32 K, bf16 row-major
  __shared__ unsigned int Bs32[64 * 16];   // 64 n x 32 K, bf16 n-major (B^T)

  const int tid  = threadIdx.x;
  const int lane = tid & 31;
  const int wave = tid >> 5;               // 8 row tiles of 16
  const int ln   = lane & 15;
  const int hf   = lane >> 4;

  const int m0 = blockIdx.x * 128;
  const int n0 = blockIdx.y * 64;

  f32x8 c[4] = {};

  for (int k0 = 0; k0 < K; k0 += 32) {
    // ---- stage A tile (128x32) as packed bf16 pairs; idx == row*16 + kp ----
#pragma unroll
    for (int i = 0; i < 8; ++i) {
      int idx = tid + i * 256;             // 0..2047 u32 slots
      int row = idx >> 4;
      int kp  = idx & 15;
      int mg  = m0 + row;
      unsigned int packed = 0u;
      if (mg < M_ROWS) {
        if (A_BF16) {
          const unsigned short* A16 = (const unsigned short*)Aptr;
          packed = *(const unsigned int*)(A16 + (size_t)mg * K + k0 + 2 * kp);
        } else {
          const float* A32 = (const float*)Aptr;
          float f0 = A32[(size_t)mg * K + k0 + 2 * kp];
          float f1 = A32[(size_t)mg * K + k0 + 2 * kp + 1];
          packed = (unsigned int)f2bf(f0) | ((unsigned int)f2bf(f1) << 16);
        }
      }
      As32[idx] = packed;
    }
    // ---- stage B tile (32x64), transposed into [n][k] ----
#pragma unroll
    for (int i = 0; i < 8; ++i) {
      int idx = i * 256 + tid;             // coalesced in n
      int nn  = idx & 63;
      int kk  = idx >> 6;
      float f = Bmat[(size_t)(k0 + kk) * N + (n0 + nn)];
      ((unsigned short*)Bs32)[nn * 32 + kk] = f2bf(f);
    }
    __syncthreads();

    u32x8 af;
#pragma unroll
    for (int v = 0; v < 8; ++v)
      af[v] = As32[(wave * 16 + ln) * 16 + (a_kpair(v, hf) >> 1)];
#pragma unroll
    for (int t = 0; t < 4; ++t) {
      u32x8 bf;
#pragma unroll
      for (int v = 0; v < 8; ++v) {
        int k = a_kpair(v, hf);
        bf[v] = Bs32[(t * 16 + ln) * 16 + (k >> 1)];
      }
      c[t] = wmma_bf16(af, bf, c[t]);
    }
    __syncthreads();
  }

  // ---- epilogue: bias + store ----
#pragma unroll
  for (int t = 0; t < 4; ++t) {
    int n_g = n0 + t * 16 + ln;
    float bv = bias[n_g];
#pragma unroll
    for (int r = 0; r < 8; ++r) {
      int m_g = m0 + wave * 16 + r + 8 * hf;
      if (m_g < M_ROWS) {
        float val = c[t][r] + bv;
        if (QKV_EPI) {
          // channel c = j*12 + h (interleaved head split of the reference)
          int hh = n_g % N_HEAD;
          int j  = n_g / N_HEAD;
          int bb = m_g / L_SEQ;
          int li = m_g % L_SEQ;
          size_t base = ((size_t)(bb * N_HEAD + hh) * L_PAD + li) * D_HEAD;
          unsigned short bfv = f2bf(val);
          if      (j <     D_HEAD) Qb[base + j]              = bfv;
          else if (j < 2 * D_HEAD) Kb[base + j -     D_HEAD] = bfv;
          else                     Vb[base + j - 2 * D_HEAD] = bfv;
        } else {
          Cout[(size_t)m_g * N + n_g] = val;
        }
      }
    }
  }
}

// ---------------------------------------------------------------------------
// Flash attention: one block per (b, h, 128-row chunk); 8 waves x 16 rows.
// K tile staged with GLOBAL_LOAD_ASYNC_TO_LDS_B128 (ASYNCcnt); row max via
// ds_swizzle xor-reduction; row sums via WMMA against an all-ones fragment.
// ---------------------------------------------------------------------------
__global__ __launch_bounds__(256) void attn_flash_wmma(
    const unsigned short* __restrict__ Qb,
    const unsigned short* __restrict__ Kb,
    const unsigned short* __restrict__ Vb,
    unsigned short* __restrict__ Ab)
{
  __shared__ unsigned int Ks32[32 * 32];    // 32 keys x 64 d, bf16  [key][d]
  __shared__ unsigned int Vs32[64 * 16];    // 64 d x 32 keys, bf16  [d][key]
  __shared__ unsigned int Ps32[8][16 * 16]; // per-wave 16x32 bf16 P bounce

  const int tid  = threadIdx.x;
  const int lane = tid & 31;
  const int wave = tid >> 5;
  const int ln   = lane & 15;
  const int hf   = lane >> 4;

  const int bh = blockIdx.y;
  const int b  = bh / N_HEAD;
  const int h  = bh % N_HEAD;
  const int m0 = blockIdx.x * 128 + wave * 16;
  const size_t head = (size_t)bh * L_PAD * D_HEAD;

  // Q fragments: rows m0..m0+15, two K-chunks of 32 over d_head.
  u32x8 qa[2];
  {
    int li = m0 + ln; if (li > L_SEQ - 1) li = L_SEQ - 1;
    const unsigned short* qrow = Qb + head + (size_t)li * D_HEAD;
#pragma unroll
    for (int f = 0; f < 2; ++f)
#pragma unroll
      for (int v = 0; v < 8; ++v)
        qa[f][v] = *(const unsigned int*)(qrow + f * 32 + a_kpair(v, hf));
  }

  u32x8 ones;
#pragma unroll
  for (int v = 0; v < 8; ++v) ones[v] = 0x3F803F80u;   // bf16 1.0 pairs

  f32x8 o[4] = {};
  float m_run[8], l_run[8];
#pragma unroll
  for (int r = 0; r < 8; ++r) { m_run[r] = -3.0e38f; l_run[r] = 0.0f; }

  const int NIT = L_PAD / 32;               // 33 full key-tiles

  for (int it = 0; it < NIT; ++it) {
    const int n0 = it * 32;
    __syncthreads();
    // ---- K tile: contiguous 4KB, async global->LDS (16B per lane) ----
    {
      unsigned lds = (unsigned)(size_t)(const void*)((const char*)Ks32 + tid * 16);
      unsigned long long g =
          (unsigned long long)(size_t)(Kb + head + (size_t)n0 * D_HEAD) + tid * 16;
      async_load_b128(lds, g);
    }
    // ---- V tile transposed [d][key] (manual: async cannot transpose) ----
#pragma unroll
    for (int i = 0; i < 4; ++i) {
      int idx = tid + i * 256;
      int key = idx >> 5;
      int dp  = idx & 31;
      unsigned int u =
          *(const unsigned int*)(Vb + head + (size_t)(n0 + key) * D_HEAD + 2 * dp);
      unsigned short* V16 = (unsigned short*)Vs32;
      V16[(2 * dp)     * 32 + key] = (unsigned short)(u & 0xFFFFu);
      V16[(2 * dp + 1) * 32 + key] = (unsigned short)(u >> 16);
    }
    // prefetch next K tile into cache (global_prefetch_b8)
    if (it + 1 < NIT)
      __builtin_prefetch(Kb + head + (size_t)(n0 + 32) * D_HEAD + (tid & 255) * 8, 0, 1);
    wait_asynccnt0();                       // this wave's async copies done
    __syncthreads();                        // all waves' tiles visible

    // ---- scores: S = Q @ K^T  (M=16, N=2x16 keys, K=64 over d_head) ----
    f32x8 s[2];
#pragma unroll
    for (int nt = 0; nt < 2; ++nt) {
      f32x8 cc = {};
#pragma unroll
      for (int kf = 0; kf < 2; ++kf) {
        u32x8 bk;
#pragma unroll
        for (int v = 0; v < 8; ++v) {
          int dd = kf * 32 + a_kpair(v, hf);
          bk[v] = Ks32[(nt * 16 + ln) * 32 + (dd >> 1)];
        }
        cc = wmma_bf16(qa[kf], bk, cc);
      }
      s[nt] = cc;
    }
    // ---- scale 1/sqrt(64) and pad-column mask ----
#pragma unroll
    for (int nt = 0; nt < 2; ++nt) {
      bool oob = (n0 + nt * 16 + ln) >= L_SEQ;
#pragma unroll
      for (int r = 0; r < 8; ++r)
        s[nt][r] = oob ? -3.0e38f : s[nt][r] * 0.125f;
    }
    // ---- row max: ds_swizzle xor-reduce over the 16 lanes of each half ----
    float alpha[8];
#pragma unroll
    for (int r = 0; r < 8; ++r) {
      float t = fmaxf(s[0][r], s[1][r]);
      t = fmaxf(t, swz_xor<1>(t));
      t = fmaxf(t, swz_xor<2>(t));
      t = fmaxf(t, swz_xor<4>(t));
      t = fmaxf(t, swz_xor<8>(t));
      float mn = fmaxf(m_run[r], t);
      alpha[r] = __expf(m_run[r] - mn);
      m_run[r] = mn;
    }
#pragma unroll
    for (int nt = 0; nt < 2; ++nt)
#pragma unroll
      for (int r = 0; r < 8; ++r)
        s[nt][r] = __expf(s[nt][r] - m_run[r]);
    // ---- rescale running O by alpha ----
#pragma unroll
    for (int r = 0; r < 8; ++r)
#pragma unroll
      for (int t4 = 0; t4 < 4; ++t4) o[t4][r] *= alpha[r];

    // ---- P: C-layout -> A-layout via per-wave LDS bounce ----
    // Same-wave DS ops are in-order (ISA 7.3); only block compiler reordering.
    unsigned short* P16 = (unsigned short*)Ps32[wave];
#pragma unroll
    for (int nt = 0; nt < 2; ++nt)
#pragma unroll
      for (int r = 0; r < 8; ++r)
        P16[(r + 8 * hf) * 32 + nt * 16 + ln] = f2bf(s[nt][r]);
    asm volatile("" ::: "memory");
    u32x8 pa;
#pragma unroll
    for (int v = 0; v < 8; ++v)
      pa[v] = Ps32[wave][ln * 16 + (a_kpair(v, hf) >> 1)];

    // ---- row sums on the matrix pipe: rs = P @ ones ----
    f32x8 rs = {};
    rs = wmma_bf16(pa, ones, rs);
#pragma unroll
    for (int r = 0; r < 8; ++r)
      l_run[r] = l_run[r] * alpha[r] + rs[r];

    // ---- O += P @ V  (N = 4x16 d_head cols, K = 32 keys) ----
#pragma unroll
    for (int t4 = 0; t4 < 4; ++t4) {
      u32x8 bv;
#pragma unroll
      for (int v = 0; v < 8; ++v) {
        int k = a_kpair(v, hf);
        bv[v] = Vs32[(t4 * 16 + ln) * 16 + (k >> 1)];
      }
      o[t4] = wmma_bf16(pa, bv, o[t4]);
    }
  }

  // ---- normalize and store bf16, head-slowest channel = h*64 + d ----
#pragma unroll
  for (int t4 = 0; t4 < 4; ++t4)
#pragma unroll
    for (int r = 0; r < 8; ++r) {
      int li = m0 + r + 8 * hf;
      if (li < L_SEQ) {
        float ov = o[t4][r] / l_run[r];
        Ab[((size_t)b * L_SEQ + li) * DMODEL + h * D_HEAD + t4 * 16 + ln] = f2bf(ov);
      }
    }
}

// ---------------------------------------------------------------------------
extern "C" void kernel_launch(void* const* d_in, const int* in_sizes, int n_in,
                              void* d_out, int out_size, void* d_ws, size_t ws_size,
                              hipStream_t stream) {
  const float* query = (const float*)d_in[0];
  const float* Wqkv  = (const float*)d_in[1];
  const float* bqkv  = (const float*)d_in[2];
  const float* Wo    = (const float*)d_in[3];
  const float* bo    = (const float*)d_in[4];
  float* out = (float*)d_out;

  const size_t HEADBUF = (size_t)B_SZ * N_HEAD * L_PAD * D_HEAD * sizeof(unsigned short);
  char* ws = (char*)d_ws;
  unsigned short* Qb = (unsigned short*)(ws);
  unsigned short* Kb = (unsigned short*)(ws + HEADBUF);
  unsigned short* Vb = (unsigned short*)(ws + 2 * HEADBUF);
  unsigned short* Ab = (unsigned short*)(ws + 3 * HEADBUF);

  dim3 blk(256);

  dim3 g1((M_ROWS + 127) / 128, N_QKV / 64);
  gemm_bias_wmma<false, true><<<g1, blk, 0, stream>>>(
      query, Wqkv, bqkv, nullptr, Qb, Kb, Vb, N_QKV, DMODEL);

  dim3 g2((L_SEQ + 127) / 128, B_SZ * N_HEAD);
  attn_flash_wmma<<<g2, blk, 0, stream>>>(Qb, Kb, Vb, Ab);

  dim3 g3((M_ROWS + 127) / 128, DMODEL / 64);
  gemm_bias_wmma<true, false><<<g3, blk, 0, stream>>>(
      Ab, Wo, bo, out, nullptr, nullptr, nullptr, DMODEL, DMODEL);
}
